// TrainableMMSEModule_6700148982150
// MI455X (gfx1250) — matrix operand order repaired and checked by reading, not compiled, not run on hardware
//
#include <hip/hip_runtime.h>
#include <hip/hip_bf16.h>
#include <math.h>

// ---------------------------------------------------------------------------
// Sizes from the reference
// ---------------------------------------------------------------------------
#define SEQ   8192
#define H2    4096
#define HID   2048
#define NMAT  12
#define REAL_SIZE 78     // N + N*(N-1)/2
#define IMAG_SIZE 66     // N*(N-1)/2
#define OUTF  144        // REAL_SIZE + IMAG_SIZE
#define EPS_LN 1e-5f
#define EPS_PD 1e-6f

typedef float  fvec4 __attribute__((ext_vector_type(4)));
typedef __attribute__((ext_vector_type(16))) _Float16 v16h;
typedef __attribute__((ext_vector_type(8)))  float    v8f;

// Exact GCC-vector type used by the async-to-LDS builtin prototype:
//   int __attribute__((vector_size(16)))
typedef int ivec4g __attribute__((vector_size(16)));

#define GLOBAL_AS __attribute__((address_space(1)))
#define LDS_AS    __attribute__((address_space(3)))

// ---------------------------------------------------------------------------
// GEMV: y[r] = dot(W[r,:], x) + b[r]   (optional leaky relu)
// One row per wave32, 8 waves (256 threads) per block, x staged in LDS.
// Weights are streamed with non-temporal loads (each byte used once;
// 450 MB total > 192 MB L2, so NT is the right temporal hint).
// The LDS fill uses the CDNA5 async global->LDS engine (ASYNCcnt) when the
// builtin is available, bypassing VGPRs entirely.
// ---------------------------------------------------------------------------
__global__ __launch_bounds__(256) void gemv_kernel(
    const float* __restrict__ Wc, const float* __restrict__ Wr,
    const float* __restrict__ bc, const float* __restrict__ br,
    const float* __restrict__ x,  int xWhichStride,
    float* __restrict__ y, int rows, int cols, int act)
{
    const int which = blockIdx.y;
    const float* __restrict__ W  = which ? Wr : Wc;
    const float* __restrict__ b  = which ? br : bc;
    const float* __restrict__ xx = x + (size_t)which * xWhichStride;
    float* __restrict__ yy       = y + (size_t)which * rows;

    __shared__ float xs[SEQ];            // 32 KB, max activation length

#if __has_builtin(__builtin_amdgcn_global_load_async_to_lds_b128)
    // Async DMA of the activation vector into LDS: 16B per lane per issue,
    // tracked by ASYNCcnt — no VGPR staging, load pipe stays free for weights.
    for (int i = threadIdx.x * 4; i < cols; i += 256 * 4) {
        __builtin_amdgcn_global_load_async_to_lds_b128(
            (GLOBAL_AS ivec4g*)(xx + i),
            (LDS_AS    ivec4g*)(xs + i), 0, 0);
    }
    asm volatile("s_wait_asynccnt 0x0" ::: "memory");
#else
    for (int i = threadIdx.x * 4; i < cols; i += 256 * 4) {
        *(fvec4*)(xs + i) = *(const fvec4*)(xx + i);
    }
#endif
    __syncthreads();

    const int wave = threadIdx.x >> 5;
    const int lane = threadIdx.x & 31;
    const int row  = blockIdx.x * 8 + wave;
    if (row < rows) {
        const fvec4* __restrict__ w4 = (const fvec4*)(W + (size_t)row * cols);
        const int n4 = cols >> 2;
        float acc = 0.0f;
        for (int c = lane; c < n4; c += 32) {
            fvec4 w  = __builtin_nontemporal_load(&w4[c]);   // NT: stream, don't cache
            fvec4 xv = *(const fvec4*)(xs + c * 4);
            acc += w.x * xv.x + w.y * xv.y + w.z * xv.z + w.w * xv.w;
        }
        // wave32 cross-lane reduction
        for (int off = 16; off > 0; off >>= 1)
            acc += __shfl_xor(acc, off, 32);
        if (lane == 0) {
            float v = acc + b[row];
            if (act) v = v > 0.0f ? v : 0.1f * v;
            yy[row] = v;
        }
    }
}

// ---------------------------------------------------------------------------
// LayerNorm + leakyReLU over n=4096 elements; one block per MLP.
// ---------------------------------------------------------------------------
__global__ __launch_bounds__(256) void ln_lrelu_kernel(
    const float* __restrict__ y,
    const float* __restrict__ gC, const float* __restrict__ btC,
    const float* __restrict__ gR, const float* __restrict__ btR,
    float* __restrict__ a, int n)
{
    const int which = blockIdx.x;
    const float* __restrict__ yy = y + (size_t)which * n;
    float* __restrict__ aa       = a + (size_t)which * n;
    const float* __restrict__ g  = which ? gR  : gC;
    const float* __restrict__ bt = which ? btR : btC;

    __shared__ float red[256];
    __shared__ float stats[2];
    const int tid = threadIdx.x;

    float s = 0.0f, s2 = 0.0f;
    for (int i = tid; i < n; i += 256) {
        float v = yy[i];
        s += v; s2 += v * v;
    }
    red[tid] = s; __syncthreads();
    for (int o = 128; o > 0; o >>= 1) {
        if (tid < o) red[tid] += red[tid + o];
        __syncthreads();
    }
    if (tid == 0) stats[0] = red[0] / (float)n;
    __syncthreads();
    red[tid] = s2; __syncthreads();
    for (int o = 128; o > 0; o >>= 1) {
        if (tid < o) red[tid] += red[tid + o];
        __syncthreads();
    }
    if (tid == 0) {
        float mu = stats[0];
        stats[1] = rsqrtf(red[0] / (float)n - mu * mu + EPS_LN);
    }
    __syncthreads();
    const float mu = stats[0], rs = stats[1];
    for (int i = tid; i < n; i += 256) {
        float v = (yy[i] - mu) * rs * g[i] + bt[i];
        aa[i] = v > 0.0f ? v : 0.1f * v;
    }
}

// ---------------------------------------------------------------------------
// Cholesky outer products via WMMA.
// One wave32 per MLP (EXEC all-1s, required by WMMA).
// Cr = Lr Lr^T + Li Li^T + eps I ; Ci = Li Lr^T - Lr Li^T, 12x12 padded to 16.
// Uses V_WMMA_F32_16X16X32_F16 (K padded to 32 with zeros).
// A layout (16-bit A 16x32, ISA 7.12.2): lane L -> M = L%16, element i
//   k = (i>=8 ? 16:0) + (L>=16 ? 8:0) + (i&7)
// B layout (32x16): lane L -> N = L%16, element i -> k = (i>=8 ? 16:0) + (L>=16 ? 8:0) + (i&7)
// D layout (f32 16x16): lane L -> N = L%16, vgpr r -> M = r + (L>=16 ? 8:0)
// ---------------------------------------------------------------------------
__device__ __forceinline__ float softplus_f(float v)
{
    return v > 0.0f ? v + log1pf(expf(-v)) : log1pf(expf(v));
}

__global__ __launch_bounds__(32) void chol_wmma_kernel(
    const float* __restrict__ y4, float* __restrict__ out)
{
    const int which = blockIdx.x;
    const float* __restrict__ f = y4 + which * OUTF;

    const int lane = threadIdx.x;
    const int lo = lane & 15;
    const int hi = lane >> 4;

    // Lr(r,c): lower-tri with softplus diag; Li(r,c): strict lower-tri
    auto getLr = [&](int r, int c) -> float {
        if (r >= NMAT || c >= NMAT || c > r) return 0.0f;
        float v = f[r * (r + 1) / 2 + c];
        return (c == r) ? softplus_f(v) : v;
    };
    auto getLi = [&](int r, int c) -> float {
        if (r >= NMAT || c >= NMAT || c >= r) return 0.0f;
        return f[REAL_SIZE + r * (r - 1) / 2 + c];
    };

    v16h aLr, aLi, bLrT, bLiT;
    #pragma unroll
    for (int i = 0; i < 16; ++i) {
        int ka = ((i >= 8) ? 16 : 0) + (hi ? 8 : 0) + (i & 7);
        aLr[i] = (_Float16)getLr(lo, ka);
        aLi[i] = (_Float16)getLi(lo, ka);
        int kb = ((i >= 8) ? 16 : 0) + (hi ? 8 : 0) + (i & 7);
        bLrT[i] = (_Float16)getLr(lo, kb);   // B[k][n] = Lr[n][k]
        bLiT[i] = (_Float16)getLi(lo, kb);   // B[k][n] = Li[n][k]
    }

    v8f cr = {};
    cr = __builtin_amdgcn_wmma_f32_16x16x32_f16(false, aLr, false, bLrT,
                                                (short)0, cr, false, false);
    cr = __builtin_amdgcn_wmma_f32_16x16x32_f16(false, aLi, false, bLiT,
                                                (short)0, cr, false, false);
    v8f ci = {};
    ci = __builtin_amdgcn_wmma_f32_16x16x32_f16(false, aLi, false, bLrT,
                                                (short)0, ci, false, false);
    v8f t = {};
    t  = __builtin_amdgcn_wmma_f32_16x16x32_f16(false, aLr, false, bLiT,
                                                (short)0, t, false, false);

    float* __restrict__ o = out + which * 2 * OUTF;
    #pragma unroll
    for (int r = 0; r < 8; ++r) {
        int m = r + (hi ? 8 : 0);
        int n = lo;
        if (m < NMAT && n < NMAT) {
            float crv = cr[r] + ((m == n) ? EPS_PD : 0.0f);
            float civ = ci[r] - t[r];
            o[m * NMAT + n]        = crv;
            o[OUTF + m * NMAT + n] = civ;
        }
    }
}

// ---------------------------------------------------------------------------
// Host-side launch
// ---------------------------------------------------------------------------
extern "C" void kernel_launch(void* const* d_in, const int* in_sizes, int n_in,
                              void* d_out, int out_size, void* d_ws, size_t ws_size,
                              hipStream_t stream)
{
    const float* x = (const float*)d_in[0];

    // Parameter leaves per MLP, in reference order:
    // W1 b1 g1 bt1 W2 b2 g2 bt2 W3 b3 W4 b4
    const float* P[2][12];
    if (n_in >= 25) {
        for (int w = 0; w < 2; ++w)
            for (int i = 0; i < 12; ++i)
                P[w][i] = (const float*)d_in[1 + w * 12 + i];
    } else {
        // Tuple passed as one concatenated blob per MLP
        static const size_t off[12] = {
            0ull,
            33554432ull,            // b1
            33558528ull,            // g1
            33562624ull,            // bt1
            33566720ull,            // W2
            50343936ull,            // b2
            50348032ull,            // g2
            50352128ull,            // bt2
            50356224ull,            // W3
            58744832ull,            // b3
            58746880ull,            // W4
            59041792ull             // b4
        };
        for (int w = 0; w < 2; ++w) {
            const float* base = (const float*)d_in[1 + w];
            for (int i = 0; i < 12; ++i) P[w][i] = base + off[i];
        }
    }

    // Workspace (floats): y1[2*4096] a1[2*4096] y2[2*4096] a2[2*4096]
    //                     a3[2*2048] y4[2*144]   -> ~149 KB
    float* ws = (float*)d_ws;
    float* y1 = ws;
    float* a1 = ws + 2 * H2;
    float* y2 = ws + 4 * H2;
    float* a2 = ws + 6 * H2;
    float* a3 = ws + 8 * H2;
    float* y4 = ws + 8 * H2 + 2 * HID;

    dim3 blk(256);

    // L1: 4096 x 8192 GEMV (x shared between MLPs -> stride 0)
    gemv_kernel<<<dim3(H2 / 8, 2), blk, 0, stream>>>(
        P[0][0], P[1][0], P[0][1], P[1][1], x, 0, y1, H2, SEQ, 0);
    ln_lrelu_kernel<<<dim3(2), blk, 0, stream>>>(
        y1, P[0][2], P[0][3], P[1][2], P[1][3], a1, H2);

    // L2: 4096 x 4096
    gemv_kernel<<<dim3(H2 / 8, 2), blk, 0, stream>>>(
        P[0][4], P[1][4], P[0][5], P[1][5], a1, H2, y2, H2, H2, 0);
    ln_lrelu_kernel<<<dim3(2), blk, 0, stream>>>(
        y2, P[0][6], P[0][7], P[1][6], P[1][7], a2, H2);

    // L3: 2048 x 4096, fused leaky relu
    gemv_kernel<<<dim3(HID / 8, 2), blk, 0, stream>>>(
        P[0][8], P[1][8], P[0][9], P[1][9], a2, H2, a3, HID, H2, 1);

    // L4: 144 x 2048
    gemv_kernel<<<dim3(OUTF / 8, 2), blk, 0, stream>>>(
        P[0][10], P[1][10], P[0][11], P[1][11], a3, HID, y4, OUTF, HID, 0);

    // Cholesky outer products -> d_out (2,2,12,12) via WMMA
    chol_wmma_kernel<<<dim3(2), dim3(32), 0, stream>>>(y4, (float*)d_out);
}
